// GigaNetEncoder_22368189677942
// MI455X (gfx1250) — matrix operand order repaired and checked by reading, not compiled, not run on hardware
//
#include <hip/hip_runtime.h>

// ---------------------------------------------------------------------------
// GigaNetEncoder for MI455X (gfx1250, wave32).
// Dense GEMMs: v_wmma_f32_16x16x32_f16, B staged through double-buffered LDS
// with global_load_async_to_lds_b128 (ASYNCcnt), 4 WMMAs per A-fragment.
// Irregular graph ops (segment softmax/sum) are f32 atomic scatter kernels.
// ---------------------------------------------------------------------------

typedef _Float16 f16;
typedef _Float16 v16h __attribute__((ext_vector_type(16)));
typedef _Float16 v8h  __attribute__((ext_vector_type(8)));
typedef float    v8f  __attribute__((ext_vector_type(8)));

#define DEVFN static __device__ __forceinline__

namespace {
constexpr int   kNA  = 600;           // agents
constexpr int   kTT  = 50;            // timesteps
constexpr int   kM   = kNA * kTT;     // 30000 nodes (multiple of 16)
constexpr float kPI  = 3.14159265358979323846f;
constexpr float kTwoPI = 6.28318530717958647692f;
constexpr int   FLAG_ACC = 1;
}

DEVFN float s_atan2(float y, float x) {
  if (x == 0.f && y == 0.f) return 0.f;
  return atan2f(y, x);
}
DEVFN float s_norm(float x, float y) {
  float sq = x * x + y * y;
  return sq > 0.f ? sqrtf(sq) : 0.f;
}
DEVFN float wrap_pi(float a) {
  float r = fmodf(a + kPI, kTwoPI);
  if (r < 0.f) r += kTwoPI;
  return r - kPI;
}
DEVFN void atomic_max_f32(float* addr, float val) {
  unsigned* ua = (unsigned*)addr;
  unsigned cur = __float_as_uint(*addr);
  while (__uint_as_float(cur) < val) {
    unsigned prev = atomicCAS(ua, cur, __float_as_uint(val));
    if (prev == cur) break;
    cur = prev;
  }
}

// --------------------------- elementwise utils -----------------------------

__global__ void k_fill(float* p, float v, int n) {
  int i = blockIdx.x * blockDim.x + threadIdx.x;
  if (i < n) p[i] = v;
}
__global__ void k_f2h(const float* __restrict__ a, f16* __restrict__ b, int n) {
  int i = blockIdx.x * blockDim.x + threadIdx.x;
  if (i < n) b[i] = (f16)a[i];
}
__global__ void k_relu_f2h(const float* __restrict__ a, f16* __restrict__ b, int n) {
  int i = blockIdx.x * blockDim.x + threadIdx.x;
  if (i < n) { float v = a[i]; b[i] = (f16)(v > 0.f ? v : 0.f); }
}

// --------------------------- feature kernels -------------------------------

// head / heading-vector / node features, temporal node order m = n*T + t
__global__ void k_feat(const float* __restrict__ pos, float* __restrict__ head,
                       float* __restrict__ hv, float* __restrict__ featx, int M) {
  int m = blockIdx.x * blockDim.x + threadIdx.x;
  if (m >= M) return;
  int t = m % kTT;
  float px = pos[2 * m], py = pos[2 * m + 1];
  float mx = 0.f, my = 0.f;
  if (t > 0) { mx = px - pos[2 * (m - 1)]; my = py - pos[2 * (m - 1) + 1]; }
  float h = s_atan2(my, mx);
  float cx = cosf(h), cy = sinf(h);
  head[m] = h; hv[2 * m] = cx; hv[2 * m + 1] = cy;
  featx[2 * m] = s_norm(mx, my);
  featx[2 * m + 1] = s_atan2(cx * my - cy * mx, cx * mx + cy * my);
}

// temporal edge features (4 dims)
__global__ void k_efeat_t(const float* __restrict__ pos, const float* __restrict__ head,
                          const float* __restrict__ hv, const int* __restrict__ ei,
                          int E, int Ep, float* __restrict__ feat) {
  int e = blockIdx.x * blockDim.x + threadIdx.x;
  if (e >= Ep) return;
  if (e >= E) { feat[4*e] = feat[4*e+1] = feat[4*e+2] = feat[4*e+3] = 0.f; return; }
  int s = ei[e], d = ei[E + e];
  float rx = pos[2*s] - pos[2*d], ry = pos[2*s+1] - pos[2*d+1];
  float cx = hv[2*d], cy = hv[2*d+1];
  feat[4*e]   = s_norm(rx, ry);
  feat[4*e+1] = s_atan2(cx * ry - cy * rx, cx * rx + cy * ry);
  feat[4*e+2] = wrap_pi(head[s] - head[d]);
  feat[4*e+3] = (float)(s - d);
}

// spatial edge features (3 dims); spatial index s = t*N + n -> temporal n*T + t
__global__ void k_efeat_s(const float* __restrict__ pos, const float* __restrict__ head,
                          const float* __restrict__ hv, const int* __restrict__ ei,
                          int E, int Ep, float* __restrict__ feat) {
  int e = blockIdx.x * blockDim.x + threadIdx.x;
  if (e >= Ep) return;
  if (e >= E) { feat[3*e] = feat[3*e+1] = feat[3*e+2] = 0.f; return; }
  int ss = ei[e], ds = ei[E + e];
  int s = (ss % kNA) * kTT + (ss / kNA);
  int d = (ds % kNA) * kTT + (ds / kNA);
  float rx = pos[2*s] - pos[2*d], ry = pos[2*s+1] - pos[2*d+1];
  float cx = hv[2*d], cy = hv[2*d+1];
  feat[3*e]   = s_norm(rx, ry);
  feat[3*e+1] = s_atan2(cx * ry - cy * rx, cx * rx + cy * ry);
  feat[3*e+2] = wrap_pi(head[s] - head[d]);
}

// Fourier features z = [cos(2*pi*x*f), sin(2*pi*x*f), x, pad] -> f16, K padded 129->160
__global__ void k_z(const float* __restrict__ feat, const float* __restrict__ freqs,
                    int d, int din, int rows, int rowsP, f16* __restrict__ z) {
  int i = blockIdx.x * blockDim.x + threadIdx.x;
  if (i >= rowsP * 160) return;
  int m = i / 160, kk = i - m * 160;
  float x = (m < rows) ? feat[m * din + d] : 0.f;
  float v;
  if (kk < 64)        v = cosf(kTwoPI * x * freqs[d * 64 + kk]);
  else if (kk < 128)  v = sinf(kTwoPI * x * freqs[d * 64 + kk - 64]);
  else if (kk == 128) v = x;
  else                v = 0.f;
  z[(size_t)m * 160 + kk] = (f16)v;
}

// --------------------------- weight packing --------------------------------
// W is row-major (N, Ksrc)  (i.e. out = x @ W.T).  Pack into wave32 B-fragment
// layout: dst[((kt*ntn + nt)*32 + lane)*16 + j] with n = nt*16 + (lane&15),
// k = kt*32 + (lane>>4)*16 + j.  Zero-fill k >= Ksrc.
__global__ void k_pack(const float* __restrict__ W, f16* __restrict__ dst,
                       int N, int Ksrc, int Kpad) {
  int idx = blockIdx.x * blockDim.x + threadIdx.x;
  if (idx >= Kpad * N) return;
  int j = idx & 15;
  int lane = (idx >> 4) & 31;
  int t = idx >> 9;
  int ntn = N >> 4;
  int nt = t % ntn, kt = t / ntn;
  int n = (nt << 4) + (lane & 15);
  int k = (kt << 5) + ((lane >> 4) << 4) + j;
  dst[idx] = (k < Ksrc) ? (f16)W[(size_t)n * Ksrc + k] : (f16)0.f;
}

// --------------------------- WMMA GEMM -------------------------------------
// C[rowsP, N] = A[rowsP, Kpad](f16 row-major) * Bpacked + bias (+C if ACC).
// Block = 8 waves: 128-row M-strip x 64-col N-group. The 4KB packed-B slab per
// K-step is shared by all waves -> staged via global_load_async_to_lds_b128
// into double-buffered LDS (ASYNCcnt + barrier). Each wave holds one A
// fragment and issues 4 WMMAs (one per 16-col tile) per K-step.
__global__ void __launch_bounds__(256)
k_gemm(const f16* __restrict__ A, const f16* __restrict__ Bp,
       const float* __restrict__ bias, float* __restrict__ C,
       int tilesM, int N, int Kpad, int flags) {
  __shared__ _Float16 ldsb[2][2048];   // 2 x 4KB B stage
  int tilesN = N >> 4;
  int ngroups = tilesN >> 2;           // N is 128 or 512 -> multiple of 64
  int ng = blockIdx.x % ngroups;
  int strip = blockIdx.x / ngroups;
  int waveId = threadIdx.x >> 5;
  int lane = threadIdx.x & 31;
  int tid = threadIdx.x;
  int mt = strip * 8 + waveId;
  bool active = (mt < tilesM);         // wave-uniform: EXEC stays all-1s
  int ktiles = Kpad >> 5;
  int hi = lane >> 4, lm = lane & 15;
  const v8h* Arow = (const v8h*)(A + (size_t)(((active ? mt : 0) << 4) + lm) * Kpad);
  const char* Bbase = (const char*)Bp;

  // async-stage one 4KB B slab (K-step kt, this block's N-group) into LDS buf
  auto stage = [&](int kt, int bufi) {
    const char* src = Bbase + (size_t)(kt * tilesN + (ng << 2)) * 1024 + tid * 16;
    unsigned ldsa = (unsigned)(unsigned long long)(const void*)(&ldsb[bufi][0]) + tid * 16;
    asm volatile("global_load_async_to_lds_b128 %0, %1, off"
                 :: "v"(ldsa), "v"((unsigned long long)src) : "memory");
  };

  v8f z8 = {0.f, 0.f, 0.f, 0.f, 0.f, 0.f, 0.f, 0.f};
  v8f acc[4] = {z8, z8, z8, z8};
  union { v16h v; v8h h[2]; } a;

  stage(0, 0);
  asm volatile("s_wait_asynccnt 0x0" ::: "memory");
  __syncthreads();

  for (int kt = 0; kt < ktiles; ++kt) {
    int cb = kt & 1;
    if (kt + 1 < ktiles) stage(kt + 1, cb ^ 1);
    if (active) {
      // 16-bit A fragment: halves 0-15 <- K = kt*32 + hi*16 .. +15
      a.h[0] = Arow[(kt << 2) + (hi << 1)];
      a.h[1] = Arow[(kt << 2) + (hi << 1) + 1];
      if (kt + 1 < ktiles)
        __builtin_prefetch((const void*)(Arow + ((kt + 1) << 2) + hi), 0, 1);
      const v16h* Bl = (const v16h*)&ldsb[cb][0];
#pragma unroll
      for (int nt = 0; nt < 4; ++nt) {
        v16h bf = Bl[(nt << 5) + lane];
        acc[nt] = __builtin_amdgcn_wmma_f32_16x16x32_f16(false, a.v, false, bf,
                                                         (short)0, acc[nt], false, false);
      }
    }
    asm volatile("s_wait_asynccnt 0x0" ::: "memory");
    __syncthreads();
  }

  if (!active) return;
  int m0 = (mt << 4) + (hi << 3);      // C/D layout: vgpr r -> row m0 + r
#pragma unroll
  for (int nt = 0; nt < 4; ++nt) {
    int n0 = ((ng << 2) + nt) * 16 + lm;
    float bv = bias ? bias[n0] : 0.f;
#pragma unroll
    for (int r = 0; r < 8; ++r) {
      size_t idx = (size_t)(m0 + r) * N + n0;
      float v = acc[nt][r] + bv;
      if (flags & FLAG_ACC) v += C[idx];
      C[idx] = v;
    }
  }
}

// --------------------------- LayerNorm (wave per row of 128) ---------------

__global__ void k_ln(const float* __restrict__ in, const float* __restrict__ g,
                     const float* __restrict__ b, f16* __restrict__ out,
                     int rowsP, int relu) {
  int row = blockIdx.x * 8 + (threadIdx.x >> 5);
  if (row >= rowsP) return;
  int lane = threadIdx.x & 31;
  const float* r = in + (size_t)row * 128;
  float v0 = r[lane], v1 = r[lane + 32], v2 = r[lane + 64], v3 = r[lane + 96];
  float s = v0 + v1 + v2 + v3;
  for (int o = 16; o; o >>= 1) s += __shfl_xor(s, o);
  float mu = s * (1.f / 128.f);
  float d0 = v0 - mu, d1 = v1 - mu, d2 = v2 - mu, d3 = v3 - mu;
  float q = d0 * d0 + d1 * d1 + d2 * d2 + d3 * d3;
  for (int o = 16; o; o >>= 1) q += __shfl_xor(q, o);
  float inv = rsqrtf(q * (1.f / 128.f) + 1e-5f);
  f16* w = out + (size_t)row * 128;
  float dv[4] = {d0, d1, d2, d3};
#pragma unroll
  for (int i = 0; i < 4; ++i) {
    int c = lane + 32 * i;
    float o2 = dv[i] * inv * g[c] + b[c];
    if (relu && o2 < 0.f) o2 = 0.f;
    w[c] = (f16)o2;
  }
}

// --------------------------- edge attention --------------------------------

__global__ void k_edge_sim(const float* __restrict__ qv, const float* __restrict__ kv,
                           const float* __restrict__ krv, const int* __restrict__ ei,
                           int E, float* __restrict__ sim, float* __restrict__ mmax) {
  int t = blockIdx.x * blockDim.x + threadIdx.x;
  if (t >= E * 8) return;
  int e = t >> 3, h = t & 7;
  int s = ei[e], d = ei[E + e];
  const float* qp = qv + (size_t)d * 128 + h * 16;
  const float* kp = kv + (size_t)s * 128 + h * 16;
  const float* rp = krv + (size_t)e * 128 + h * 16;
  float acc = 0.f;
#pragma unroll
  for (int c = 0; c < 16; ++c) acc += qp[c] * (kp[c] + rp[c]);
  acc *= 0.25f;   // DH^-0.5, DH = 16
  sim[t] = acc;
  atomic_max_f32(&mmax[d * 8 + h], acc);
}

__global__ void k_edge_exp(float* __restrict__ sim, const float* __restrict__ mmax,
                           const int* __restrict__ ei, int E, float* __restrict__ den) {
  int t = blockIdx.x * blockDim.x + threadIdx.x;
  if (t >= E * 8) return;
  int e = t >> 3, h = t & 7;
  int d = ei[E + e];
  float v = expf(sim[t] - mmax[d * 8 + h]);
  sim[t] = v;
  atomicAdd(&den[d * 8 + h], v);
}

__global__ void k_edge_agg(const float* __restrict__ sim, const float* __restrict__ den,
                           const float* __restrict__ vv, const float* __restrict__ vrv,
                           const int* __restrict__ ei, int E, float* __restrict__ agg) {
  int t = blockIdx.x * blockDim.x + threadIdx.x;
  if (t >= E * 128) return;
  int e = t >> 7, c = t & 127, h = c >> 4;
  int s = ei[e], d = ei[E + e];
  float a = sim[e * 8 + h] / den[d * 8 + h];
  atomicAdd(&agg[(size_t)d * 128 + c],
            (vv[(size_t)s * 128 + c] + vrv[(size_t)e * 128 + c]) * a);
}

__global__ void k_concat(const f16* __restrict__ a, const f16* __restrict__ b,
                         f16* __restrict__ o, int M) {
  int t = blockIdx.x * blockDim.x + threadIdx.x;
  if (t >= M * 256) return;
  int m = t >> 8, c = t & 255;
  o[t] = (c < 128) ? a[(size_t)m * 128 + c] : b[(size_t)m * 128 + c - 128];
}

__global__ void k_gate(float* __restrict__ agg, const float* __restrict__ graw,
                       const float* __restrict__ sv, int n) {
  int t = blockIdx.x * blockDim.x + threadIdx.x;
  if (t >= n) return;
  float g = 1.f / (1.f + expf(-graw[t]));
  agg[t] = agg[t] + g * (sv[t] - agg[t]);
}

// node-order permute: temporal (n*T+t) <-> spatial (t*N+n)
__global__ void k_perm(const float* __restrict__ in, float* __restrict__ out, int toSpatial) {
  int t = blockIdx.x * blockDim.x + threadIdx.x;
  if (t >= kM * 128) return;
  int row = t >> 7, c = t & 127;
  int src;
  if (toSpatial) { int tt = row / kNA, n = row % kNA; src = n * kTT + tt; }
  else           { int n = row / kTT, tt = row % kTT; src = tt * kNA + n; }
  out[t] = in[(size_t)src * 128 + c];
}

// ---------------------------------------------------------------------------
// Host orchestration
// ---------------------------------------------------------------------------

extern "C" void kernel_launch(void* const* d_in, const int* in_sizes, int n_in,
                              void* d_out, int out_size, void* d_ws, size_t ws_size,
                              hipStream_t stream) {
  (void)n_in; (void)out_size; (void)ws_size;

  const float* pos = (const float*)d_in[0];
  const int* ei_t = (const int*)d_in[130];
  const int* ei_s = (const int*)d_in[131];
  const int Et = in_sizes[130] / 2;
  const int Es = in_sizes[131] / 2;
  const int EpT = (Et + 15) & ~15;
  const int EpS = (Es + 15) & ~15;
  int maxEp = EpT > EpS ? EpT : EpS;
  if (kM > maxEp) maxEp = kM;

  // parameter tables (dict insertion order)
  const float* P_fe[3][11];
  for (int i = 0; i < 11; ++i) {
    P_fe[0][i] = (const float*)d_in[1 + i];    // fe_xa   (din=2)
    P_fe[1][i] = (const float*)d_in[12 + i];   // fe_rt   (din=4)
    P_fe[2][i] = (const float*)d_in[23 + i];   // fe_ra2a (din=3)
  }
  const float* P_at[4][24];
  const int bases[4] = {34, 58, 82, 106};      // attn_t[0], attn_t[1], attn_a2a[0], attn_a2a[1]
  for (int l = 0; l < 4; ++l)
    for (int i = 0; i < 24; ++i)
      P_at[l][i] = (const float*)d_in[bases[l] + i];

  // --- bump allocator over d_ws ---
  char* wsp = (char*)d_ws;
  size_t off = 0;
  auto alloc = [&](size_t bytes) -> void* {
    void* p = wsp + off;
    off = (off + bytes + 255) & ~(size_t)255;
    return p;
  };

  float* head  = (float*)alloc((size_t)kM * 4);
  float* hv    = (float*)alloc((size_t)kM * 8);
  float* featx = (float*)alloc((size_t)kM * 8);
  float* featT = (float*)alloc((size_t)EpT * 16);
  float* featS = (float*)alloc((size_t)EpS * 12);
  float* r_t   = (float*)alloc((size_t)EpT * 512);
  float* r_s   = (float*)alloc((size_t)EpS * 512);
  float* xA    = (float*)alloc((size_t)kM * 512);
  float* xB    = (float*)alloc((size_t)kM * 512);
  f16*   x16   = (f16*)alloc((size_t)kM * 256);
  f16*   rn16  = (f16*)alloc((size_t)maxEp * 256);
  float* qb    = (float*)alloc((size_t)kM * 512);
  float* kb    = (float*)alloc((size_t)kM * 512);
  float* vb    = (float*)alloc((size_t)kM * 512);
  float* krb   = (float*)alloc((size_t)maxEp * 512);
  float* vrb   = (float*)alloc((size_t)maxEp * 512);
  float* sim   = (float*)alloc((size_t)maxEp * 32);
  float* mmax  = (float*)alloc((size_t)kM * 32);
  float* den   = (float*)alloc((size_t)kM * 32);
  float* agg   = (float*)alloc((size_t)kM * 512);
  f16*   agg16 = (f16*)alloc((size_t)kM * 256);
  f16*   cat16 = (f16*)alloc((size_t)kM * 512);
  float* gb    = (float*)alloc((size_t)kM * 512);
  float* sb    = (float*)alloc((size_t)kM * 512);
  float* ffn32 = (float*)alloc((size_t)kM * 2048);
  f16*   ffn16 = (f16*)alloc((size_t)kM * 1024);
  f16*   zbuf  = (f16*)alloc((size_t)maxEp * 320);
  float* h32   = (float*)alloc((size_t)maxEp * 512);
  f16*   h16p  = (f16*)alloc((size_t)maxEp * 256);
  float* sacc  = (float*)alloc((size_t)maxEp * 512);

  auto L1 = [](int n) { return dim3((unsigned)((n + 255) / 256)); };
  auto zero = [&](float* p, int n) {
    k_fill<<<L1(n), 256, 0, stream>>>(p, 0.f, n);
  };
  auto gemm = [&](const f16* A, const f16* B, const float* bias, float* C,
                  int rowsP, int N, int Kpad, int flags) {
    int tilesM = rowsP >> 4;
    int strips = (tilesM + 7) >> 3;
    int ngroups = (N >> 4) >> 2;
    k_gemm<<<dim3((unsigned)(strips * ngroups)), dim3(256), 0, stream>>>(
        A, B, bias, C, tilesM, N, Kpad, flags);
  };
  auto packB = [&](const float* W, int N, int Ksrc, int Kpad) -> f16* {
    f16* dst = (f16*)alloc((size_t)Kpad * N * 2);
    k_pack<<<L1(Kpad * N), 256, 0, stream>>>(W, dst, N, Ksrc, Kpad);
    return dst;
  };
  auto lnrows = [](int r) { return dim3((unsigned)((r + 7) / 8)); };

  // --- pack all weights into WMMA fragment layout (deterministic, every call) ---
  f16* feW1[3][4]; f16* feW2[3][4]; f16* feWo[3];
  const int feDin[3] = {2, 4, 3};
  for (int f = 0; f < 3; ++f) {
    for (int d = 0; d < feDin[f]; ++d) {
      feW1[f][d] = packB(P_fe[f][1] + (size_t)d * 128 * 129, 128, 129, 160);
      feW2[f][d] = packB(P_fe[f][5] + (size_t)d * 128 * 128, 128, 128, 128);
    }
    feWo[f] = packB(P_fe[f][9], 128, 128, 128);
  }
  struct APack { f16 *q, *k, *v, *kr, *vr, *s, *g, *o, *f1, *f2; };
  auto packAttn = [&](const float* const* P) {
    APack a;
    a.q  = packB(P[0],  128, 128, 128);
    a.k  = packB(P[2],  128, 128, 128);
    a.v  = packB(P[3],  128, 128, 128);
    a.kr = packB(P[5],  128, 128, 128);
    a.vr = packB(P[6],  128, 128, 128);
    a.s  = packB(P[8],  128, 128, 128);
    a.g  = packB(P[10], 128, 256, 256);
    a.o  = packB(P[12], 128, 128, 128);
    a.f1 = packB(P[20], 512, 128, 128);
    a.f2 = packB(P[22], 128, 512, 512);
    return a;
  };
  APack AP[4];
  for (int l = 0; l < 4; ++l) AP[l] = packAttn(P_at[l]);

  // --- stage 0: geometric features ---
  k_feat<<<L1(kM), 256, 0, stream>>>(pos, head, hv, featx, kM);
  k_efeat_t<<<L1(EpT), 256, 0, stream>>>(pos, head, hv, ei_t, Et, EpT, featT);
  k_efeat_s<<<L1(EpS), 256, 0, stream>>>(pos, head, hv, ei_s, Es, EpS, featS);

  // --- Fourier encoder: per input-dim GEMM (rows x 160)x(160 x 128) etc. ---
  auto fourier = [&](const float* feat, int din, int rows, int rowsP,
                     const float* const* P, f16* const* W1p, f16* const* W2p,
                     f16* Wop, float* outbuf) {
    zero(sacc, rowsP * 128);
    for (int d = 0; d < din; ++d) {
      k_z<<<L1(rowsP * 160), 256, 0, stream>>>(feat, P[0], d, din, rows, rowsP, zbuf);
      gemm(zbuf, W1p[d], P[2] + (size_t)d * 128, h32, rowsP, 128, 160, 0);
      k_ln<<<lnrows(rowsP), 256, 0, stream>>>(h32, P[3] + (size_t)d * 128,
                                              P[4] + (size_t)d * 128, h16p, rowsP, 1);
      gemm(h16p, W2p[d], P[6] + (size_t)d * 128, sacc, rowsP, 128, 128, FLAG_ACC);
    }
    k_ln<<<lnrows(rowsP), 256, 0, stream>>>(sacc, P[7], P[8], h16p, rowsP, 1);
    gemm(h16p, Wop, P[10], outbuf, rowsP, 128, 128, 0);
  };

  fourier(featx, 2, kM, kM,  P_fe[0], feW1[0], feW2[0], feWo[0], xA);
  fourier(featT, 4, Et, EpT, P_fe[1], feW1[1], feW2[1], feWo[1], r_t);
  fourier(featS, 3, Es, EpS, P_fe[2], feW1[2], feW2[2], feWo[2], r_s);

  // --- graph attention block ---
  auto attn = [&](float* x, const float* r, const int* ei, int E, int Ep,
                  const float* const* P, const APack& W) {
    k_ln<<<lnrows(kM), 256, 0, stream>>>(x, P[14], P[15], x16, kM, 0);   // xn
    k_ln<<<lnrows(Ep), 256, 0, stream>>>(r, P[16], P[17], rn16, Ep, 0);  // rn
    gemm(x16,  W.q,  P[1],    qb,  kM, 128, 128, 0);
    gemm(x16,  W.k,  nullptr, kb,  kM, 128, 128, 0);
    gemm(x16,  W.v,  P[4],    vb,  kM, 128, 128, 0);
    gemm(rn16, W.kr, nullptr, krb, Ep, 128, 128, 0);
    gemm(rn16, W.vr, P[7],    vrb, Ep, 128, 128, 0);
    k_fill<<<L1(kM * 8), 256, 0, stream>>>(mmax, -3.0e38f, kM * 8);
    zero(den, kM * 8);
    k_edge_sim<<<L1(E * 8), 256, 0, stream>>>(qb, kb, krb, ei, E, sim, mmax);
    k_edge_exp<<<L1(E * 8), 256, 0, stream>>>(sim, mmax, ei, E, den);
    zero(agg, kM * 128);
    k_edge_agg<<<L1(E * 128), 256, 0, stream>>>(sim, den, vb, vrb, ei, E, agg);
    k_f2h<<<L1(kM * 128), 256, 0, stream>>>(agg, agg16, kM * 128);
    k_concat<<<L1(kM * 256), 256, 0, stream>>>(agg16, x16, cat16, kM);
    gemm(cat16, W.g, P[11], gb, kM, 128, 256, 0);
    gemm(x16,   W.s, P[9],  sb, kM, 128, 128, 0);
    k_gate<<<L1(kM * 128), 256, 0, stream>>>(agg, gb, sb, kM * 128);
    k_f2h<<<L1(kM * 128), 256, 0, stream>>>(agg, agg16, kM * 128);
    gemm(agg16, W.o, P[13], x, kM, 128, 128, FLAG_ACC);                  // x += agg @ Wo.T + bo
    k_ln<<<lnrows(kM), 256, 0, stream>>>(x, P[18], P[19], h16p, kM, 0);  // FFN
    gemm(h16p, W.f1, P[21], ffn32, kM, 512, 128, 0);
    k_relu_f2h<<<L1(kM * 512), 256, 0, stream>>>(ffn32, ffn16, kM * 512);
    gemm(ffn16, W.f2, P[23], x, kM, 128, 512, FLAG_ACC);                 // x += ffn
  };

  // --- layer loop with temporal<->spatial permutes ---
  float* cur = xA;
  float* alt = xB;
  float* outp = (float*)d_out;
  for (int layer = 0; layer < 2; ++layer) {
    attn(cur, r_t, ei_t, Et, EpT, P_at[layer], AP[layer]);           // temporal order
    k_perm<<<L1(kM * 128), 256, 0, stream>>>(cur, alt, 1);           // -> spatial
    { float* t = cur; cur = alt; alt = t; }
    attn(cur, r_s, ei_s, Es, EpS, P_at[2 + layer], AP[2 + layer]);   // spatial order
    float* dst = (layer == 1) ? outp : alt;
    k_perm<<<L1(kM * 128), 256, 0, stream>>>(cur, dst, 0);           // -> temporal
    if (layer == 0) { float* t = cur; cur = alt; alt = t; }
  }
}
